// GNN_70824010711256
// MI455X (gfx1250) — matrix operand order repaired and checked by reading, not compiled, not run on hardware
//
#include <hip/hip_runtime.h>
#include <hip/hip_bf16.h>
#include <math.h>

typedef __attribute__((ext_vector_type(16))) __bf16 v16bf;
typedef __attribute__((ext_vector_type(8)))  float  v8f;

#define NN 50000
#define DD 128
#define EE 600000

// ---------------------------------------------------------------------------
// Degree (int atomics) and dinv = rsqrt(deg + 1)
// ---------------------------------------------------------------------------
__global__ __launch_bounds__(256) void deg_kernel(const int* __restrict__ ei,
                                                  int* __restrict__ deg) {
  int e = blockIdx.x * 256 + threadIdx.x;
  if (e >= EE) return;
  atomicAdd(&deg[ei[EE + e]], 1);
}

__global__ __launch_bounds__(256) void dinv_kernel(const int* __restrict__ deg,
                                                   float* __restrict__ dinv) {
  int i = blockIdx.x * 256 + threadIdx.x;
  if (i >= NN) return;
  dinv[i] = rsqrtf((float)deg[i] + 1.0f);
}

// ---------------------------------------------------------------------------
// hw = h @ W   (bf16 split-precision WMMA, fp32 accumulate)
// Block: 256 thr = 8 waves; wave w -> rows [blk*128 + w*16, +16), cols 0..127
// ---------------------------------------------------------------------------
__global__ __launch_bounds__(256) void gemm_kernel(const float* __restrict__ h,
                                                   const float* __restrict__ W,
                                                   float* __restrict__ hw) {
  __shared__ alignas(64) __bf16 sWhi[128 * 32];  // W^T tile: [n][kk]
  __shared__ alignas(64) __bf16 sWlo[128 * 32];

  const int t       = threadIdx.x;
  const int lane    = t & 31;
  const int wave    = t >> 5;
  const int lane16  = lane & 15;
  const int halfSel = lane >> 4;                 // 0: lanes 0-15, 1: lanes 16-31
  const int m0      = blockIdx.x * 128 + wave * 16;

  int row = m0 + lane16;
  if (row >= NN) row = NN - 1;                   // clamp loads; stores guarded
  const float* __restrict__ hrow = h + (size_t)row * DD;

  v8f acc[8];
#pragma unroll
  for (int i = 0; i < 8; ++i)
#pragma unroll
    for (int j = 0; j < 8; ++j) acc[i][j] = 0.0f;

  for (int kt = 0; kt < 4; ++kt) {
    const int k0 = kt * 32;
    __syncthreads();
    // stage W^T (hi/lo bf16): sW[n*32 + kk] = W[(k0+kk)*128 + n]
    for (int i = t; i < 32 * 128; i += 256) {
      const int kk = i >> 7;
      const int n  = i & 127;
      float  w   = W[(k0 + kk) * DD + n];
      __bf16 whi = (__bf16)w;
      sWhi[n * 32 + kk] = whi;
      sWlo[n * 32 + kk] = (__bf16)(w - (float)whi);
    }
    __syncthreads();

    // A fragment: lanes 0-15 hold K={0..15}, lanes 16-31 K={16..31}
    const int kb = k0 + halfSel * 16;
    float av[16];
#pragma unroll
    for (int i = 0; i < 8; ++i) av[i]     = hrow[kb + i];
#pragma unroll
    for (int i = 0; i < 8; ++i) av[8 + i] = hrow[kb + 8 + i];
    v16bf ahi, alo;
#pragma unroll
    for (int i = 0; i < 16; ++i) {
      __bf16 hi = (__bf16)av[i];
      ahi[i] = hi;
      alo[i] = (__bf16)(av[i] - (float)hi);
    }

#pragma unroll
    for (int nt = 0; nt < 8; ++nt) {
      const int col = nt * 16 + lane16;          // B: lane = column, K packed
      const v16bf bhi = *(const v16bf*)&sWhi[col * 32 + halfSel * 16];
      const v16bf blo = *(const v16bf*)&sWlo[col * 32 + halfSel * 16];
      acc[nt] = __builtin_amdgcn_wmma_f32_16x16x32_bf16(
          false, ahi, false, bhi, (short)0, acc[nt], false, false);
      acc[nt] = __builtin_amdgcn_wmma_f32_16x16x32_bf16(
          false, ahi, false, blo, (short)0, acc[nt], false, false);
      acc[nt] = __builtin_amdgcn_wmma_f32_16x16x32_bf16(
          false, alo, false, bhi, (short)0, acc[nt], false, false);
    }
  }

  // C/D layout: VGPR r -> M = r (lanes 0-15) / 8+r (lanes 16-31), N = lane&15
#pragma unroll
  for (int nt = 0; nt < 8; ++nt) {
    const int col = nt * 16 + lane16;
#pragma unroll
    for (int r = 0; r < 8; ++r) {
      const int rr = m0 + r + halfSel * 8;
      if (rr < NN) hw[(size_t)rr * DD + col] = acc[nt][r];
    }
  }
}

// ---------------------------------------------------------------------------
// agg = hw * dinv^2 (self-loop) + bias       (one float4 per thread)
// ---------------------------------------------------------------------------
__global__ __launch_bounds__(256) void init_agg_kernel(
    const float* __restrict__ hw, const float* __restrict__ dinv,
    const float* __restrict__ b, float* __restrict__ agg) {
  int idx = blockIdx.x * 256 + threadIdx.x;     // over NN*32 float4s
  if (idx >= NN * 32) return;
  int node = idx >> 5;
  int q    = idx & 31;
  float sn = dinv[node];
  sn = sn * sn;
  float4 hv = ((const float4*)hw)[idx];
  float4 bv = ((const float4*)b)[q];
  float4 r;
  r.x = hv.x * sn + bv.x;
  r.y = hv.y * sn + bv.y;
  r.z = hv.z * sn + bv.z;
  r.w = hv.w * sn + bv.w;
  ((float4*)agg)[idx] = r;
}

// ---------------------------------------------------------------------------
// Edge scatter: one wave32 per edge, float4 per lane.
// Hardware f32 atomic forced via inline CDNA5 asm (no CAS fallback),
// device scope so the RMW executes at the coherent L2 atomic units.
// ---------------------------------------------------------------------------
__device__ __forceinline__ void atomic_addf(float* p, float v) {
  asm volatile("global_atomic_add_f32 %0, %1, off scope:SCOPE_DEV"
               :
               : "v"(p), "v"(v)
               : "memory");
}

__global__ __launch_bounds__(256) void scatter_kernel(
    const float* __restrict__ hw, const int* __restrict__ ei,
    const float* __restrict__ dinv, float* __restrict__ agg) {
  const int wid  = (blockIdx.x * 256 + threadIdx.x) >> 5;
  const int lane = threadIdx.x & 31;
  if (wid >= EE) return;
  const int   src = ei[wid];
  const int   dst = ei[EE + wid];
  const float s   = dinv[src] * dinv[dst];
  const float4 v  = ((const float4*)(hw + (size_t)src * DD))[lane];
  float* out = agg + (size_t)dst * DD + lane * 4;
  atomic_addf(out + 0, v.x * s);
  atomic_addf(out + 1, v.y * s);
  atomic_addf(out + 2, v.z * s);
  atomic_addf(out + 3, v.w * s);
}

// ---------------------------------------------------------------------------
// h += prelu(agg) (or plain add for the output layer), in place on d_out
// ---------------------------------------------------------------------------
__global__ __launch_bounds__(256) void combine_kernel(
    const float* __restrict__ agg, const float* __restrict__ alpha_p,
    int use_prelu, float* __restrict__ hio) {
  int idx = blockIdx.x * 256 + threadIdx.x;
  if (idx >= NN * 32) return;
  float4 v = ((const float4*)agg)[idx];
  if (use_prelu) {
    const float a = *alpha_p;
    v.x = v.x >= 0.0f ? v.x : a * v.x;
    v.y = v.y >= 0.0f ? v.y : a * v.y;
    v.z = v.z >= 0.0f ? v.z : a * v.z;
    v.w = v.w >= 0.0f ? v.w : a * v.w;
  }
  float4 hv = ((float4*)hio)[idx];
  hv.x += v.x; hv.y += v.y; hv.z += v.z; hv.w += v.w;
  ((float4*)hio)[idx] = hv;
}

// ---------------------------------------------------------------------------
extern "C" void kernel_launch(void* const* d_in, const int* in_sizes, int n_in,
                              void* d_out, int out_size, void* d_ws, size_t ws_size,
                              hipStream_t stream) {
  (void)in_sizes; (void)n_in; (void)out_size; (void)ws_size;
  const float* x      = (const float*)d_in[0];
  const int*   ei     = (const int*)d_in[1];
  const float* Ws     = (const float*)d_in[2];
  const float* bs     = (const float*)d_in[3];
  const float* alphas = (const float*)d_in[4];
  float* h = (float*)d_out;

  char* ws = (char*)d_ws;
  const size_t szND = (size_t)NN * DD * sizeof(float);   // 25,600,000 B
  float* dinv = (float*)ws;                              // NN floats
  float* hw   = (float*)(ws + 200192);                   // NN*DD floats
  float* agg  = (float*)(ws + 200192 + szND);            // NN*DD floats
  int*   deg  = (int*)agg;                               // reuse agg pre-loop

  // degree + dinv
  hipMemsetAsync(deg, 0, (size_t)NN * sizeof(int), stream);
  deg_kernel<<<(EE + 255) / 256, 256, 0, stream>>>(ei, deg);
  dinv_kernel<<<(NN + 255) / 256, 256, 0, stream>>>(deg, dinv);

  // h = x
  hipMemcpyAsync(h, x, szND, hipMemcpyDeviceToDevice, stream);

  const int gemm_blocks  = (NN + 127) / 128;      // 391
  const int elem4_blocks = (NN * 32 + 255) / 256; // NN*DD/4 threads
  const int edge_blocks  = (EE + 7) / 8;          // 1 wave per edge

  for (int l = 0; l < 5; ++l) {
    gemm_kernel<<<gemm_blocks, 256, 0, stream>>>(h, Ws + (size_t)l * DD * DD, hw);
    init_agg_kernel<<<elem4_blocks, 256, 0, stream>>>(hw, dinv, bs + l * DD, agg);
    scatter_kernel<<<edge_blocks, 256, 0, stream>>>(hw, ei, dinv, agg);
    const int use_prelu = (l < 4) ? 1 : 0;
    combine_kernel<<<elem4_blocks, 256, 0, stream>>>(
        agg, alphas + (l < 4 ? l : 0), use_prelu, h);
  }
}
